// GraphVAE_70935679861346
// MI455X (gfx1250) — compile-verified, matrix operand order checked by reading, not run on hardware
//
#include <hip/hip_runtime.h>
#include <hip/hip_bf16.h>

#define NN 100000
#define EE 1000000
#define GG 256
#define FIN 64
#define FHID 64
#define FOUT 32

typedef __attribute__((ext_vector_type(16))) _Float16 v16h;
typedef __attribute__((ext_vector_type(8)))  float    v8f;

// ---------------------------------------------------------------- utilities

__global__ __launch_bounds__(256) void fill_kernel(float* p, float v, int n) {
    int i = blockIdx.x * 256 + threadIdx.x;
    if (i < n) p[i] = v;
}

// deg[dst] += 1 for every edge (deg pre-initialized to 1.0 for self loops)
__global__ __launch_bounds__(256) void deg_acc_kernel(const long long* __restrict__ dst,
                                                      float* __restrict__ deg, int ne) {
    int e = blockIdx.x * 256 + threadIdx.x;
    if (e < ne) atomicAdd(&deg[(int)dst[e]], 1.0f);
}

__global__ __launch_bounds__(256) void rsqrt_kernel(float* p, int n) {
    int i = blockIdx.x * 256 + threadIdx.x;
    if (i < n) p[i] = rsqrtf(p[i]);
}

// ---------------------------------------------------------------- WMMA GEMM
// Y[n][64]  = X[n][64] @ W[64][64]        (row-major, fin x fout weights)
// AG[n][64] = Y[n][64] * dinv[n]^2        (fused GCN self-loop init)
// grid.x = n/32 blocks, 256 threads = 8 wave32; each wave owns a 16x16 tile.
__global__ __launch_bounds__(256) void gemm_n64_kernel(const float* __restrict__ X,
                                                       const float* __restrict__ W,
                                                       const float* __restrict__ dinv,
                                                       float* __restrict__ Y,
                                                       float* __restrict__ AG) {
    __shared__ alignas(16) _Float16 sW[64 * 64];   // transposed f16: sW[n*64 + k]
    __shared__ alignas(16) float    sXf[32 * 64];  // f32 tile: sXf[m*64 + k]
    const int t = threadIdx.x;
    const int row0 = blockIdx.x * 32;

    // --- async-copy the 32x64 f32 X tile into LDS (ASYNCcnt-tracked DMA) ---
    {
        unsigned lds0 = (unsigned)(uintptr_t)(&sXf[t * 4]);   // LDS byte addr (aperture low bits)
        const float* g0 = X + row0 * 64 + t * 4;
        asm volatile("global_load_async_to_lds_b128 %0, %1, off"
                     :: "v"(lds0), "v"(g0) : "memory");
        unsigned lds1 = lds0 + 4096;
        const float* g1 = g0 + 1024;
        asm volatile("global_load_async_to_lds_b128 %0, %1, off"
                     :: "v"(lds1), "v"(g1) : "memory");
    }

    // --- meanwhile stage W transposed into LDS as f16 (coalesced along fout) ---
    for (int i = t; i < 64 * 64; i += 256) {
        int k = i >> 6, n = i & 63;
        sW[n * 64 + k] = (_Float16)W[i];
    }

    asm volatile("s_wait_asynccnt 0x0" ::: "memory");
    __syncthreads();

    const int wave = t >> 5;          // 0..7
    const int lane = t & 31;
    const int tr = wave >> 2;         // 0..1 (row tile)
    const int tc = wave & 3;          // 0..3 (col tile)
    const int mrow = tr * 16 + (lane & 15);   // local A row
    const int ncol = tc * 16 + (lane & 15);   // B column
    const int kbase = (lane < 16) ? 0 : 8;    // K-group base per 16-bit ISA layout
    const int kb2   = kbase >> 1;

    const float*    xr   = &sXf[mrow * 64];
    const unsigned* sw32 = (const unsigned*)sW;   // f16 pair view

    v8f acc = {};
#pragma unroll
    for (int kt = 0; kt < 2; ++kt) {              // K = 64 -> 2 x (16x16x32)
        union { v16h v; unsigned u[8]; } A, B;
#pragma unroll
        for (int j = 0; j < 8; ++j) {
            int k = kt * 32 + ((j < 4) ? (kbase + 2 * j) : (16 + kbase + 2 * (j - 4)));
            A.v[2 * j]     = (_Float16)xr[k];        // f32 LDS -> f16 fragment
            A.v[2 * j + 1] = (_Float16)xr[k + 1];
            B.u[j] = sw32[ncol * 32 + (k >> 1)];
            (void)kb2;
        }
        acc = __builtin_amdgcn_wmma_f32_16x16x32_f16(
            false, A.v, false, B.v, (short)0, acc, false, false);
    }

    // C/D layout: VGPR r -> M=r (lanes 0-15), M=r+8 (lanes 16-31); N = lane&15
    const int gr0 = row0 + tr * 16 + ((lane < 16) ? 0 : 8);
    const int gc  = tc * 16 + (lane & 15);
#pragma unroll
    for (int r = 0; r < 8; ++r) {
        float h  = acc[r];
        float di = dinv[gr0 + r];
        Y[(gr0 + r) * 64 + gc]  = h;
        AG[(gr0 + r) * 64 + gc] = h * di * di;    // fused self-loop term
    }
}

// ---------------------------------------------------------------- GCN pieces

// one wave32 per edge: agg[dst] += h[src] * dinv[src]*dinv[dst]   (64 floats)
__global__ __launch_bounds__(256) void edge_agg_kernel(const long long* __restrict__ src,
                                                       const long long* __restrict__ dst,
                                                       const float* __restrict__ h,
                                                       const float* __restrict__ dinv,
                                                       float* __restrict__ agg, int ne) {
    int e = blockIdx.x * 8 + (threadIdx.x >> 5);
    e = __builtin_amdgcn_readfirstlane(e);        // wave-uniform -> scalar loads
    if (e >= ne) return;
    int lane = threadIdx.x & 31;
    // stream-ahead prefetch of the edge index arrays (global_prefetch_b8)
    __builtin_prefetch(src + e + 2048, 0, 0);
    __builtin_prefetch(dst + e + 2048, 0, 0);
    int s = (int)src[e], d = (int)dst[e];
    float norm = dinv[s] * dinv[d];
    float2 v = ((const float2*)(h + s * 64))[lane];
    float* ap = agg + d * 64 + lane * 2;
    atomicAdd(ap,     v.x * norm);
    atomicAdd(ap + 1, v.y * norm);
}

// out[i][f] = relu(agg[i][f] + b[f])
__global__ __launch_bounds__(256) void bias_relu_kernel(const float* __restrict__ agg,
                                                        const float* __restrict__ b,
                                                        float* __restrict__ out) {
    int i = blockIdx.x * 256 + threadIdx.x;
    if (i >= NN * 64) return;
    out[i] = fmaxf(agg[i] + b[i & 63], 0.0f);
}

// one wave32 per node: sums[batch[n]] += h[n]; counts[batch[n]] += 1
__global__ __launch_bounds__(256) void pool_kernel(const long long* __restrict__ batch,
                                                   const float* __restrict__ h,
                                                   float* __restrict__ sums,
                                                   float* __restrict__ counts) {
    int n = blockIdx.x * 8 + (threadIdx.x >> 5);
    n = __builtin_amdgcn_readfirstlane(n);
    if (n >= NN) return;
    int lane = threadIdx.x & 31;
    int g = (int)batch[n];
    float2 v = ((const float2*)(h + n * 64))[lane];
    float* sp = sums + g * 64 + lane * 2;
    atomicAdd(sp,     v.x);
    atomicAdd(sp + 1, v.y);
    if (lane == 0) atomicAdd(&counts[g], 1.0f);
}

// ---------------------------------------------------------------- RNG (threefry2x32 + erfinv)

__device__ __forceinline__ unsigned rotl32(unsigned x, int d) {
    return (x << d) | (x >> (32 - d));
}

__device__ void threefry2x32(unsigned k0, unsigned k1, unsigned c0, unsigned c1,
                             unsigned* o0, unsigned* o1) {
    unsigned ks0 = k0, ks1 = k1, ks2 = 0x1BD11BDAu ^ k0 ^ k1;
    unsigned x0 = c0 + ks0, x1 = c1 + ks1;
    const int ra[4] = {13, 15, 26, 6}, rb[4] = {17, 29, 16, 24};
#pragma unroll
    for (int g = 0; g < 5; ++g) {
        const int* r = (g & 1) ? rb : ra;
#pragma unroll
        for (int i = 0; i < 4; ++i) { x0 += x1; x1 = rotl32(x1, r[i]); x1 ^= x0; }
        switch (g) {
            case 0: x0 += ks1; x1 += ks2 + 1u; break;
            case 1: x0 += ks2; x1 += ks0 + 2u; break;
            case 2: x0 += ks0; x1 += ks1 + 3u; break;
            case 3: x0 += ks1; x1 += ks2 + 4u; break;
            case 4: x0 += ks2; x1 += ks0 + 5u; break;
        }
    }
    *o0 = x0; *o1 = x1;
}

__device__ float erfinv_f(float x) {
    float w = -logf((1.0f - x) * (1.0f + x)), p;
    if (w < 5.0f) {
        w -= 2.5f;
        p = 2.81022636e-08f;
        p = 3.43273939e-07f + p * w;
        p = -3.5233877e-06f + p * w;
        p = -4.39150654e-06f + p * w;
        p = 0.00021858087f + p * w;
        p = -0.00125372503f + p * w;
        p = -0.00417768164f + p * w;
        p = 0.246640727f + p * w;
        p = 1.50140941f + p * w;
    } else {
        w = sqrtf(w) - 3.0f;
        p = -0.000200214257f;
        p = 0.000100950558f + p * w;
        p = 0.00134934322f + p * w;
        p = -0.00367342844f + p * w;
        p = 0.00573950773f + p * w;
        p = -0.0076224613f + p * w;
        p = 0.00943887047f + p * w;
        p = 1.00167406f + p * w;
        p = 2.83297682f + p * w;
    }
    return p * x;
}

__device__ float jax_normal42(int idx) {   // idx in [0, 8192)
    const int half = (GG * FOUT) / 2;      // 4096
    int j = (idx < half) ? idx : idx - half;
    unsigned o0, o1;
    threefry2x32(0u, 42u, (unsigned)j, (unsigned)(j + half), &o0, &o1);
    unsigned bits = (idx < half) ? o0 : o1;
    float u01 = __uint_as_float((bits >> 9) | 0x3f800000u) - 1.0f;   // [0,1)
    const float lo = -0.99999994f;                                    // nextafter(-1,0)
    float u = u01 * (1.0f - lo) + lo;
    return 1.41421356f * erfinv_f(u);
}

// ---------------------------------------------------------------- VAE head + decoder

__global__ __launch_bounds__(64) void heads_kernel(const float* __restrict__ sums,
                                                   const float* __restrict__ counts,
                                                   const float* __restrict__ Wmu,
                                                   const float* __restrict__ bmu,
                                                   const float* __restrict__ Wlv,
                                                   const float* __restrict__ blv,
                                                   float* __restrict__ mu_out,
                                                   float* __restrict__ lv_out,
                                                   float* __restrict__ z) {
    __shared__ float p[64];
    __shared__ float smu[32], slv[32];
    int g = blockIdx.x, t = threadIdx.x;
    float cnt = fmaxf(counts[g], 1.0f);
    p[t] = sums[g * 64 + t] / cnt;
    __syncthreads();
    int o = t & 31;
    const float* W = (t < 32) ? Wmu : Wlv;
    const float* b = (t < 32) ? bmu : blv;
    float acc = b[o];
#pragma unroll 8
    for (int k = 0; k < 64; ++k) acc += p[k] * W[k * 32 + o];
    if (t < 32) smu[o] = acc; else slv[o] = acc;
    __syncthreads();
    if (t < 32) {
        float m = smu[o], lv = slv[o];
        mu_out[g * 32 + o] = m;
        lv_out[g * 32 + o] = lv;
        float eps = jax_normal42(g * 32 + o);
        z[g * 32 + o] = m + eps * expf(0.5f * lv);
    }
}

__global__ __launch_bounds__(64) void decoder_kernel(const float* __restrict__ z,
                                                     const float* __restrict__ Wd1,
                                                     const float* __restrict__ bd1,
                                                     const float* __restrict__ Wd2,
                                                     const float* __restrict__ bd2,
                                                     float* __restrict__ recon) {
    __shared__ float zz[32];
    __shared__ float hid[64];
    int g = blockIdx.x, t = threadIdx.x;
    if (t < 32) zz[t] = z[g * 32 + t];
    __syncthreads();
    float a = bd1[t];
#pragma unroll 8
    for (int k = 0; k < 32; ++k) a += zz[k] * Wd1[k * 64 + t];
    hid[t] = fmaxf(a, 0.0f);
    __syncthreads();
    float o = bd2[t];
#pragma unroll 8
    for (int k = 0; k < 64; ++k) o += hid[k] * Wd2[k * 64 + t];
    recon[g * 64 + t] = o;
}

// ---------------------------------------------------------------- launch

extern "C" void kernel_launch(void* const* d_in, const int* in_sizes, int n_in,
                              void* d_out, int out_size, void* d_ws, size_t ws_size,
                              hipStream_t stream) {
    const float*     x    = (const float*)d_in[0];
    const long long* ei   = (const long long*)d_in[1];   // [2, E] int64
    const long long* bat  = (const long long*)d_in[2];   // [N] int64
    const float* W1  = (const float*)d_in[3];
    const float* b1  = (const float*)d_in[4];
    const float* W2  = (const float*)d_in[5];
    const float* b2  = (const float*)d_in[6];
    const float* Wmu = (const float*)d_in[7];
    const float* bmu = (const float*)d_in[8];
    const float* Wlv = (const float*)d_in[9];
    const float* blv = (const float*)d_in[10];
    const float* Wd1 = (const float*)d_in[11];
    const float* bd1 = (const float*)d_in[12];
    const float* Wd2 = (const float*)d_in[13];
    const float* bd2 = (const float*)d_in[14];

    const long long* src = ei;
    const long long* dst = ei + EE;

    float* out = (float*)d_out;
    float* recon  = out;                        // [G,64]
    float* mu_out = out + GG * FIN;             // [G,32]
    float* lv_out = mu_out + GG * FOUT;         // [G,32]

    float* ws     = (float*)d_ws;
    float* dinv   = ws;                         // [N]   (deg -> rsqrt in place)
    float* bufA   = dinv + NN;                  // [N,64]  h
    float* bufB   = bufA + NN * 64;             // [N,64]  agg / activations
    float* sums   = bufB + NN * 64;             // [G,64]
    float* counts = sums + GG * 64;             // [G]
    float* zbuf   = counts + GG;                // [G,32]

    const int elemBlocks = (NN * 64 + 255) / 256;   // 25000
    const int gemmBlocks = NN / 32;                 // 3125 (exact)
    const int edgeBlocks = EE / 8;                  // 125000 (exact)
    const int waveBlocks = (NN + 7) / 8;            // 12500

    // degrees -> dinv
    fill_kernel<<<(NN + 255) / 256, 256, 0, stream>>>(dinv, 1.0f, NN);
    deg_acc_kernel<<<(EE + 255) / 256, 256, 0, stream>>>(dst, dinv, EE);
    rsqrt_kernel<<<(NN + 255) / 256, 256, 0, stream>>>(dinv, NN);

    // layer 1: h = x@W1 (+ fused agg init) ; edges ; bufB = relu(agg + b1)
    gemm_n64_kernel<<<gemmBlocks, 256, 0, stream>>>(x, W1, dinv, bufA, bufB);
    edge_agg_kernel<<<edgeBlocks, 256, 0, stream>>>(src, dst, bufA, dinv, bufB, EE);
    bias_relu_kernel<<<elemBlocks, 256, 0, stream>>>(bufB, b1, bufB);

    // layer 2 (gemm reads bufB then the agg init overwrites it next kernel-step safe:
    // gemm writes bufA=h and needs a fresh agg target -> reuse bufB after read)
    gemm_n64_kernel<<<gemmBlocks, 256, 0, stream>>>(bufB, W2, dinv, bufA, bufB);
    edge_agg_kernel<<<edgeBlocks, 256, 0, stream>>>(src, dst, bufA, dinv, bufB, EE);
    bias_relu_kernel<<<elemBlocks, 256, 0, stream>>>(bufB, b2, bufB);

    // global mean pool
    fill_kernel<<<(GG * 64 + GG + 255) / 256, 256, 0, stream>>>(sums, 0.0f, GG * 64 + GG);
    pool_kernel<<<waveBlocks, 256, 0, stream>>>(bat, bufB, sums, counts);

    // heads + reparameterize + decode
    heads_kernel<<<GG, 64, 0, stream>>>(sums, counts, Wmu, bmu, Wlv, blv,
                                        mu_out, lv_out, zbuf);
    decoder_kernel<<<GG, 64, 0, stream>>>(zbuf, Wd1, bd1, Wd2, bd2, recon);
}
